// Seq2SeqUltimate_39204461477932
// MI455X (gfx1250) — matrix operand order repaired and checked by reading, not compiled, not run on hardware
//
#include <hip/hip_runtime.h>
#include <hip/hip_bf16.h>
#include <cstddef>

// ---------------------------------------------------------------------------
// Types for CDNA5 WMMA (gfx1250, wave32)
// ---------------------------------------------------------------------------
typedef _Float16 half_t;
typedef __attribute__((ext_vector_type(16))) _Float16 v16h;
typedef __attribute__((ext_vector_type(8)))  float    v8f;

union FragU { uint4 u4[2]; v16h v; };

// A-matrix 16x32 f16 fragment (ISA 7.12.2): lane m = lane&15,
// K-chunks: [kg*8 .. kg*8+7] and [16+kg*8 .. 16+kg*8+7] with kg = lane>>4.
__device__ __forceinline__ v16h load_fragA(const half_t* A, int lda, int row0,
                                           int k0, int lane) {
  const half_t* p = A + (size_t)(row0 + (lane & 15)) * lda + k0 + ((lane >> 4) << 3);
  FragU f;
  f.u4[0] = *reinterpret_cast<const uint4*>(p);
  f.u4[1] = *reinterpret_cast<const uint4*>(p + 16);
  return f.v;
}
// B-matrix 32x16 f16 fragment: lane n = lane&15, K-chunks
// [kg*8 .. kg*8+7] and [16+kg*8 .. 16+kg*8+7].  Loading rows of a row-major [N,K] weight gives Wᵀ.
__device__ __forceinline__ v16h load_fragB(const half_t* W, int ldw, int col0,
                                           int k0, int lane) {
  const half_t* p = W + (size_t)(col0 + (lane & 15)) * ldw + k0 + ((lane >> 4) << 3);
  FragU f;
  f.u4[0] = *reinterpret_cast<const uint4*>(p);
  f.u4[1] = *reinterpret_cast<const uint4*>(p + 16);
  return f.v;
}

__device__ __forceinline__ float sigmoidf_(float x) {
  return 1.0f / (1.0f + __expf(-x));
}

// ---------------------------------------------------------------------------
// Generic WMMA GEMM:  C[M,N] = act( A[M,K] * W[N,K]^T + bias + bias2 )  (beta=0)
//                     C[M,N] = C + A*W^T                                (beta=1)
// Optional f16 mirrors Cbf/Cbf2 (for feeding later WMMA GEMMs).
// One 16x16 tile per wave; 256-thread blocks = 8 wave32 waves.
// ---------------------------------------------------------------------------
__global__ void k_gemm(const half_t* __restrict__ A, int lda,
                       const half_t* __restrict__ W, int ldw,
                       const float* __restrict__ bias,
                       const float* __restrict__ bias2,
                       float* __restrict__ C, int ldc,
                       half_t* __restrict__ Cbf, int ldcbf,
                       half_t* __restrict__ Cbf2, int ldcbf2,
                       int M, int N, int K, int beta, int act) {
  const int lane = threadIdx.x & 31;
  const int wid  = (blockIdx.x * blockDim.x + threadIdx.x) >> 5;
  const int Mt = M >> 4, Nt = N >> 4;
  const int tm = wid % Mt, tn = wid / Mt;
  if (tn >= Nt) return;
  const int row0 = tm << 4, col0 = tn << 4;
  const int n = col0 + (lane & 15);
  const int mBase = (lane >> 4) << 3;

  v8f acc;
  if (beta) {
#pragma unroll
    for (int r = 0; r < 8; ++r)
      acc[r] = C[(size_t)(row0 + mBase + r) * ldc + n];
  } else {
    float bv = (bias ? bias[n] : 0.0f) + (bias2 ? bias2[n] : 0.0f);
#pragma unroll
    for (int r = 0; r < 8; ++r) acc[r] = bv;
  }

  for (int k0 = 0; k0 < K; k0 += 32) {
    if (k0 + 32 < K)  // gfx1250: lowers to global_prefetch_b8
      __builtin_prefetch(A + (size_t)row0 * lda + k0 + 32, 0, 1);
    v16h a = load_fragA(A, lda, row0, k0, lane);
    v16h b = load_fragB(W, ldw, col0, k0, lane);
    acc = __builtin_amdgcn_wmma_f32_16x16x32_f16(false, a, false, b,
                                                 (short)0, acc, false, false);
  }

#pragma unroll
  for (int r = 0; r < 8; ++r) {
    float v = acc[r];
    if (act == 1) v = fmaxf(v, 0.0f);
    else if (act == 2) v = tanhf(v);
    const size_t rowi = (size_t)(row0 + mBase + r);
    C[rowi * ldc + n] = v;
    if (Cbf)  Cbf [rowi * ldcbf  + n] = (half_t)v;
    if (Cbf2) Cbf2[rowi * ldcbf2 + n] = (half_t)v;
  }
}

// ---------------------------------------------------------------------------
// Persistent single-WGP LSTM scan over S=256 steps. 1024 threads = 32 waves.
// h (f16 ping/pong) + c (f32) live entirely in the 320KB CDNA5 LDS.
// Each wave owns four (b-tile, j-tile) pairs; per pair it holds 4 gate
// accumulators (i,f,g,o) so the cell nonlinearity fuses on WMMA registers.
// ---------------------------------------------------------------------------
__global__ void k_lstm_seq(const float* __restrict__ xproj,  // [S*64, 2048]
                           const half_t* __restrict__ Whh,   // [2048, 512] f16
                           half_t* __restrict__ ybf,         // [S*64,1024] (+dirOff)
                           half_t* __restrict__ hfin,        // [64,1024]
                           half_t* __restrict__ cfin,        // [64,1024]
                           int dirOff, int fwd) {
  extern __shared__ char smem[];
  half_t* hbuf0 = (half_t*)smem;            // 64*512 f16 = 64KB
  half_t* hbuf1 = hbuf0 + 64 * 512;         // 64KB
  float*  cst   = (float*)(hbuf1 + 64 * 512);  // 64*512 f32 = 128KB
  const int tid = threadIdx.x, lane = tid & 31, wv = tid >> 5;

  for (int i = tid; i < 64 * 512; i += 1024) {
    hbuf0[i] = (half_t)0.0f;
    cst[i]   = 0.0f;
  }
  __syncthreads();

  half_t* hcur = hbuf0;
  half_t* hnxt = hbuf1;

  for (int step = 0; step < 256; ++step) {
    const int s = fwd ? step : (255 - step);
    const float* xp = xproj + (size_t)s * 64 * 2048;
#pragma unroll
    for (int p = 0; p < 4; ++p) {
      const int pairIdx = wv * 4 + p;          // 0..127
      const int b0 = (pairIdx & 3) << 4;       // 4 row tiles of 16 (B=64)
      const int j0 = (pairIdx >> 2) << 4;      // 32 col tiles of 16 (H=512)
      const int jn = j0 + (lane & 15);
      const int mBase = (lane >> 4) << 3;

      v8f acc[4];
#pragma unroll
      for (int g = 0; g < 4; ++g)
#pragma unroll
        for (int r = 0; r < 8; ++r)
          acc[g][r] = xp[(size_t)(b0 + mBase + r) * 2048 + (g << 9) + jn];

      for (int k0 = 0; k0 < 512; k0 += 32) {
        v16h a = load_fragA(hcur, 512, b0, k0, lane);
#pragma unroll
        for (int g = 0; g < 4; ++g) {
          v16h bfr = load_fragB(Whh, 512, (g << 9) + j0, k0, lane);
          acc[g] = __builtin_amdgcn_wmma_f32_16x16x32_f16(
              false, a, false, bfr, (short)0, acc[g], false, false);
        }
      }
      // fused LSTM cell (torch gate order i,f,g,o)
#pragma unroll
      for (int r = 0; r < 8; ++r) {
        const int b = b0 + mBase + r;
        const int ci = b * 512 + jn;
        const float ig = sigmoidf_(acc[0][r]);
        const float fg = sigmoidf_(acc[1][r]);
        const float gg = tanhf(acc[2][r]);
        const float og = sigmoidf_(acc[3][r]);
        const float cn = fg * cst[ci] + ig * gg;
        const float hn = og * tanhf(cn);
        cst[ci]  = cn;
        hnxt[ci] = (half_t)hn;
        ybf[(size_t)(s * 64 + b) * 1024 + dirOff + jn] = (half_t)hn;
        if (step == 255) {
          hfin[b * 1024 + dirOff + jn] = (half_t)hn;
          cfin[b * 1024 + dirOff + jn] = (half_t)cn;
        }
      }
    }
    __syncthreads();
    half_t* t = hcur; hcur = hnxt; hnxt = t;
  }
}

// ---------------------------------------------------------------------------
// Misc small kernels
// ---------------------------------------------------------------------------
__global__ void k_f32_to_f16(const float* __restrict__ s,
                             half_t* __restrict__ d, int n) {
  int i = blockIdx.x * blockDim.x + threadIdx.x;
  if (i < n) d[i] = (half_t)s[i];
}

// emb + sinusoidal PE -> f16 [S*B, 256], row = s*64+b
__global__ void k_enc_embed(const int* __restrict__ src,
                            const float* __restrict__ emb,
                            half_t* __restrict__ x) {
  int idx = blockIdx.x * blockDim.x + threadIdx.x;  // 16384*256
  int e = idx & 255, sb = idx >> 8;
  int b = sb & 63, s = sb >> 6;
  int tok = src[b * 256 + s];
  float di = (float)((e >> 1) << 1);
  float dv = __expf(di * (-9.210340371976184f / 256.0f));  // -ln(10000)/E
  float ang = (float)s * dv;
  float pe = (e & 1) ? __cosf(ang) : __sinf(ang);
  x[(size_t)(s * 64 + b) * 256 + e] = (half_t)(emb[tok * 256 + e] + pe);
}

// decoder token embedding + PE[0] (sin0=0, cos0=1) -> xin[:,0:256], comb[:,1536:1792]
__global__ void k_dec_embed(const int* __restrict__ tgt,
                            const float* __restrict__ demb, int t,
                            half_t* __restrict__ xin,
                            half_t* __restrict__ comb) {
  int idx = blockIdx.x * blockDim.x + threadIdx.x;  // 64*256
  int b = idx >> 8, e = idx & 255;
  int tok = tgt[b * 128 + t];
  float v = demb[tok * 256 + e] + ((e & 1) ? 1.0f : 0.0f);
  xin[b * 1280 + e] = (half_t)v;
  comb[b * 1792 + 1536 + e] = (half_t)v;
}

// replicate h0/c0 into the 3 decoder layers
__global__ void k_init_state(const half_t* __restrict__ h0,
                             const float* __restrict__ c0,
                             half_t* __restrict__ hdec,
                             float* __restrict__ cdec) {
  int i = blockIdx.x * blockDim.x + threadIdx.x;  // 3*32768
  int j = i % 32768;
  hdec[i] = h0[j];
  cdec[i] = c0[j];
}

__global__ void k_zero_out(float* __restrict__ out) {
  int i = blockIdx.x * blockDim.x + threadIdx.x;  // 64*6000
  int b = i / 6000, c = i % 6000;
  out[(size_t)b * 768000 + c] = 0.0f;
}

// decoder LSTM cell pointwise (gates precomputed by two GEMM passes)
__global__ void k_cell(const float* __restrict__ gates,  // [64,2048]
                       float* __restrict__ c,            // [64,512]
                       half_t* __restrict__ hbf,         // [64,512]
                       half_t* __restrict__ comb) {      // nullable [64,1792]
  int idx = blockIdx.x * blockDim.x + threadIdx.x;  // 32768
  int b = idx >> 9, j = idx & 511;
  const float* g = gates + (size_t)b * 2048;
  float ig = sigmoidf_(g[j]);
  float fg = sigmoidf_(g[512 + j]);
  float gg = tanhf(g[1024 + j]);
  float og = sigmoidf_(g[1536 + j]);
  float cn = fg * c[idx] + ig * gg;
  float hn = og * tanhf(cn);
  c[idx] = cn;
  hbf[idx] = (half_t)hn;
  if (comb) comb[b * 1792 + j] = (half_t)hn;
}

// multi-head dot-product attention: one wave32 per (b,h) pair
__global__ void k_attn(const float* __restrict__ q,   // [64,512]
                       const float* __restrict__ Kb,  // [S*64,512]
                       const float* __restrict__ Vb,  // [S*64,512]
                       half_t* __restrict__ ctxh) {   // [64,512]
  __shared__ float qS[8][64];
  __shared__ float aS[8][256];
  const int lane = threadIdx.x & 31, w = threadIdx.x >> 5;
  const int pair = blockIdx.x * 8 + w;  // 512 pairs
  const int b = pair >> 3, h = pair & 7;
  const float* qp = q + b * 512 + h * 64;
  qS[w][lane]      = qp[lane];
  qS[w][lane + 32] = qp[lane + 32];
  __syncthreads();

  float sc[8];
  float mx = -1e30f;
#pragma unroll
  for (int i = 0; i < 8; ++i) {
    int s = i * 32 + lane;
    const float* kp = Kb + ((size_t)s * 64 + b) * 512 + h * 64;
    float dot = 0.0f;
    for (int d = 0; d < 64; ++d) dot += qS[w][d] * kp[d];
    sc[i] = dot * 0.125f;  // 1/sqrt(64)
    mx = fmaxf(mx, sc[i]);
  }
  for (int m = 16; m; m >>= 1) mx = fmaxf(mx, __shfl_xor(mx, m, 32));
  float sum = 0.0f;
#pragma unroll
  for (int i = 0; i < 8; ++i) {
    sc[i] = __expf(sc[i] - mx);
    sum += sc[i];
  }
  for (int m = 16; m; m >>= 1) sum += __shfl_xor(sum, m, 32);
#pragma unroll
  for (int i = 0; i < 8; ++i) aS[w][i * 32 + lane] = sc[i];
  __syncthreads();

  const float inv = 1.0f / sum;
#pragma unroll
  for (int dd = 0; dd < 2; ++dd) {
    int d = lane + dd * 32;
    float acc = 0.0f;
    for (int s = 0; s < 256; ++s)
      acc += aS[w][s] * Vb[((size_t)s * 64 + b) * 512 + h * 64 + d];
    ctxh[b * 512 + h * 64 + d] = (half_t)(acc * inv);
  }
}

// layernorm over 512, one wave per row
__global__ void k_layernorm(const float* __restrict__ z,
                            const float* __restrict__ gamma,
                            const float* __restrict__ beta,
                            half_t* __restrict__ zn) {
  const int lane = threadIdx.x & 31, w = threadIdx.x >> 5;
  const int row = blockIdx.x * 8 + w;  // 64 rows
  const float* zr = z + (size_t)row * 512;
  float s = 0.0f, s2 = 0.0f;
  for (int i = lane; i < 512; i += 32) {
    float v = zr[i];
    s += v;
    s2 += v * v;
  }
  for (int m = 16; m; m >>= 1) {
    s  += __shfl_xor(s, m, 32);
    s2 += __shfl_xor(s2, m, 32);
  }
  float mu = s / 512.0f;
  float var = s2 / 512.0f - mu * mu;
  float iv = rsqrtf(var + 1e-5f);
  for (int i = lane; i < 512; i += 32)
    zn[(size_t)row * 512 + i] = (half_t)((zr[i] - mu) * iv * gamma[i] + beta[i]);
}

// ---------------------------------------------------------------------------
// Host orchestration
// ---------------------------------------------------------------------------
extern "C" void kernel_launch(void* const* d_in, const int* in_sizes, int n_in,
                              void* d_out, int out_size, void* d_ws,
                              size_t ws_size, hipStream_t stream) {
  (void)in_sizes; (void)n_in; (void)out_size; (void)ws_size;

  // ---- workspace bump allocator ----
  char* wsp = (char*)d_ws;
  auto alloc = [&](size_t bytes) -> void* {
    void* p = wsp;
    wsp += (bytes + 255) & ~(size_t)255;
    return p;
  };
  auto cvt = [&](int idx, size_t n) -> half_t* {
    half_t* dst = (half_t*)alloc(n * sizeof(half_t));
    int blocks = (int)((n + 255) / 256);
    k_f32_to_f16<<<blocks, 256, 0, stream>>>((const float*)d_in[idx], dst, (int)n);
    return dst;
  };
  auto gemm = [&](const half_t* A, int lda, const half_t* W, int ldw,
                  const float* bias, const float* bias2, float* C, int ldc,
                  half_t* Cbf, int ldcbf, half_t* Cbf2, int ldcbf2,
                  int M, int N, int K, int beta, int act) {
    int tiles = (M >> 4) * (N >> 4);
    int blocks = (tiles + 7) / 8;
    k_gemm<<<blocks, 256, 0, stream>>>(A, lda, W, ldw, bias, bias2, C, ldc,
                                       Cbf, ldcbf, Cbf2, ldcbf2, M, N, K, beta, act);
  };

  // ---- input index map (setup_inputs dict order) ----
  // 0 src, 1 tgt, 2 enc_emb, 3 dec_emb,
  // 4 + l*8 + d*4 + {0:Wih,1:Whh,2:bih,3:bhh}  (enc, l=0..2, d=f/b)
  // 28 fch_W, 29 fch_b, 30 fcc_W, 31 fcc_b,
  // 32 Wq, 33 bq, 34 Wk, 35 bk, 36 Wv, 37 bv, 38 Wo, 39 bo,
  // 40 + l*4 + {0:Wih,1:Whh,2:bih,3:bhh}  (dec, l=0..2)
  // 52 fc1_W, 53 fc1_b, 54 fc2_W, 55 fc2_b, 56 ln_g, 57 ln_b
  const int* src = (const int*)d_in[0];
  const int* tgt = (const int*)d_in[1];

  // ---- convert all weight matrices to f16 once ----
  half_t* encWih[3][2];
  half_t* encWhh[3][2];
  for (int l = 0; l < 3; ++l)
    for (int d = 0; d < 2; ++d) {
      size_t din = (l == 0) ? 256 : 1024;
      encWih[l][d] = cvt(4 + l * 8 + d * 4 + 0, (size_t)2048 * din);
      encWhh[l][d] = cvt(4 + l * 8 + d * 4 + 1, (size_t)2048 * 512);
    }
  half_t* fchW = cvt(28, (size_t)512 * 1024);
  half_t* fccW = cvt(30, (size_t)512 * 1024);
  half_t* Wq   = cvt(32, (size_t)512 * 512);
  half_t* Wk   = cvt(34, (size_t)512 * 1024);
  half_t* Wv   = cvt(36, (size_t)512 * 1024);
  half_t* Wo   = cvt(38, (size_t)1024 * 512);
  half_t* decWih[3];
  half_t* decWhh[3];
  for (int l = 0; l < 3; ++l) {
    size_t din = (l == 0) ? 1280 : 512;
    decWih[l] = cvt(40 + l * 4 + 0, (size_t)2048 * din);
    decWhh[l] = cvt(40 + l * 4 + 1, (size_t)2048 * 512);
  }
  half_t* fc1W = cvt(52, (size_t)512 * 1792);
  half_t* fc2W = cvt(54, (size_t)6000 * 512);

  // ---- activation buffers ----
  half_t* embh  = (half_t*)alloc((size_t)16384 * 256 * 2);
  half_t* xA    = (half_t*)alloc((size_t)16384 * 1024 * 2);
  half_t* xB    = (half_t*)alloc((size_t)16384 * 1024 * 2);
  float*  xproj = (float*)alloc((size_t)16384 * 2048 * 4);
  half_t* hfin  = (half_t*)alloc((size_t)64 * 1024 * 2);
  half_t* cfin  = (half_t*)alloc((size_t)64 * 1024 * 2);
  float*  Kbuf  = (float*)alloc((size_t)16384 * 512 * 4);
  float*  Vbuf  = (float*)alloc((size_t)16384 * 512 * 4);
  float*  h0f   = (float*)alloc((size_t)64 * 512 * 4);
  half_t* h0h   = (half_t*)alloc((size_t)64 * 512 * 2);
  float*  c0f   = (float*)alloc((size_t)64 * 512 * 4);
  half_t* hdec  = (half_t*)alloc((size_t)3 * 64 * 512 * 2);
  float*  cdec  = (float*)alloc((size_t)3 * 64 * 512 * 4);
  half_t* xin   = (half_t*)alloc((size_t)64 * 1280 * 2);
  half_t* comb  = (half_t*)alloc((size_t)64 * 1792 * 2);
  float*  qf    = (float*)alloc((size_t)64 * 512 * 4);
  half_t* ctxh  = (half_t*)alloc((size_t)64 * 512 * 2);
  float*  ctxf  = (float*)alloc((size_t)64 * 1024 * 4);
  float*  gatesf = (float*)alloc((size_t)64 * 2048 * 4);
  float*  zf    = (float*)alloc((size_t)64 * 512 * 4);
  half_t* znh   = (half_t*)alloc((size_t)64 * 512 * 2);

  // ================= Encoder =================
  k_enc_embed<<<16384, 256, 0, stream>>>(src, (const float*)d_in[2], embh);

  const half_t* Xcur = embh;
  int Kin = 256;
  half_t* Xnext = xA;
  const size_t lstmShm = 2 * 64 * 512 * sizeof(half_t) + 64 * 512 * sizeof(float);
  for (int l = 0; l < 3; ++l) {
    for (int d = 0; d < 2; ++d) {
      const float* bih = (const float*)d_in[4 + l * 8 + d * 4 + 2];
      const float* bhh = (const float*)d_in[4 + l * 8 + d * 4 + 3];
      gemm(Xcur, Kin, encWih[l][d], Kin, bih, bhh, xproj, 2048,
           nullptr, 0, nullptr, 0, 16384, 2048, Kin, 0, 0);
      k_lstm_seq<<<1, 1024, lstmShm, stream>>>(xproj, encWhh[l][d], Xnext,
                                               hfin, cfin, d * 512, d == 0 ? 1 : 0);
    }
    Xcur = Xnext;
    Kin = 1024;
    Xnext = (Xnext == xA) ? xB : xA;
  }
  const half_t* encOut = Xcur;  // [S*64, 1024] f16, row = s*64+b

  // bridge: h0 = tanh([hf,hb]@fch^T+b), c0 = tanh([cf,cb]@fcc^T+b)
  gemm(hfin, 1024, fchW, 1024, (const float*)d_in[29], nullptr, h0f, 512,
       h0h, 512, nullptr, 0, 64, 512, 1024, 0, 2);
  gemm(cfin, 1024, fccW, 1024, (const float*)d_in[31], nullptr, c0f, 512,
       nullptr, 0, nullptr, 0, 64, 512, 1024, 0, 2);
  k_init_state<<<384, 256, 0, stream>>>(h0h, c0f, hdec, cdec);

  // attention K/V precompute (row = s*64+b, col = head*64+d)
  gemm(encOut, 1024, Wk, 1024, (const float*)d_in[35], nullptr, Kbuf, 512,
       nullptr, 0, nullptr, 0, 16384, 512, 1024, 0, 0);
  gemm(encOut, 1024, Wv, 1024, (const float*)d_in[37], nullptr, Vbuf, 512,
       nullptr, 0, nullptr, 0, 16384, 512, 1024, 0, 0);

  k_zero_out<<<1500, 256, 0, stream>>>((float*)d_out);

  // ================= Decoder (127 teacher-forced steps) =================
  for (int t = 0; t < 127; ++t) {
    k_dec_embed<<<64, 256, 0, stream>>>(tgt, (const float*)d_in[3], t, xin, comb);
    // q from top decoder hidden
    gemm(hdec + 2 * 32768, 512, Wq, 512, (const float*)d_in[33], nullptr, qf, 512,
         nullptr, 0, nullptr, 0, 64, 512, 512, 0, 0);
    k_attn<<<64, 256, 0, stream>>>(qf, Kbuf, Vbuf, ctxh);
    // ctx projection -> xin[:,256:1280] and comb[:,512:1536]
    gemm(ctxh, 512, Wo, 512, (const float*)d_in[39], nullptr, ctxf, 1024,
         xin + 256, 1280, comb + 512, 1792, 64, 1024, 512, 0, 0);
    // 3-layer decoder LSTM stack
    for (int l = 0; l < 3; ++l) {
      const half_t* Ain = (l == 0) ? (const half_t*)xin : (const half_t*)(hdec + (l - 1) * 32768);
      int Kl = (l == 0) ? 1280 : 512;
      const float* bih = (const float*)d_in[40 + l * 4 + 2];
      const float* bhh = (const float*)d_in[40 + l * 4 + 3];
      gemm(Ain, Kl, decWih[l], Kl, bih, bhh, gatesf, 2048,
           nullptr, 0, nullptr, 0, 64, 2048, Kl, 0, 0);
      gemm(hdec + l * 32768, 512, decWhh[l], 512, nullptr, nullptr, gatesf, 2048,
           nullptr, 0, nullptr, 0, 64, 2048, 512, 1, 0);
      k_cell<<<128, 256, 0, stream>>>(gatesf, cdec + l * 32768, hdec + l * 32768,
                                      (l == 2) ? comb : nullptr);
    }
    // fc1 + ReLU, layernorm, fc2 -> out[:, t+1, :]
    gemm(comb, 1792, fc1W, 1792, (const float*)d_in[53], nullptr, zf, 512,
         nullptr, 0, nullptr, 0, 64, 512, 1792, 0, 1);
    k_layernorm<<<8, 256, 0, stream>>>(zf, (const float*)d_in[56],
                                       (const float*)d_in[57], znh);
    gemm(znh, 512, fc2W, 512, (const float*)d_in[55], nullptr,
         (float*)d_out + (size_t)(t + 1) * 6000, 768000,
         nullptr, 0, nullptr, 0, 64, 6000, 512, 0, 0);
  }
}
